// ExactTripletClassifier_69123203662019
// MI455X (gfx1250) — compile-verified, matrix operand order checked by reading, not run on hardware
//
#include <hip/hip_runtime.h>

// ---------------------------------------------------------------------------
// ExactTripletClassifier for MI455X (gfx1250, wave32, WMMA).
// bf16 WMMA (f32 accum) for all GEMMs; fp32 for LN / GeLU / tanh / scan.
// GEMM: block tile 256x64 (8 waves), wave tile 64x32 -> 8 wmma / K-step.
// ---------------------------------------------------------------------------

typedef __attribute__((ext_vector_type(16))) __bf16 v16bf;
typedef __attribute__((ext_vector_type(8)))  __bf16 v8bf;
typedef __attribute__((ext_vector_type(8)))  float  v8f;

#define TC_B     8
#define TC_L     2048
#define TC_D     512
#define TC_H     1024
#define TC_C     64
#define TC_R     64
#define TC_NBLK  2
#define TC_N     (TC_B * TC_L)          // 16384 rows
#define TC_NP    (TC_B * (TC_L - 1))    // 16376 prefix rows

#define LDS_PAD  40                      // bf16 elements per LDS row (80 B)
#define BLK_M    256
#define BLK_N    64

__device__ __forceinline__ int imind(int a, int b) { return a < b ? a : b; }

// ---------------------------------------------------------------------------
// Weight prep: transpose+cast f32 [K,N] -> bf16 [N,K] (B^T layout for WMMA).
// ---------------------------------------------------------------------------
__global__ void transpose_cast_kernel(__bf16* __restrict__ dst,
                                      const float* __restrict__ src,
                                      int Nrows, int Kcols) {
  int i = blockIdx.x * blockDim.x + threadIdx.x;
  if (i >= Nrows * Kcols) return;
  int n = i / Kcols;
  int k = i - n * Kcols;
  dst[i] = (__bf16)src[(size_t)k * Nrows + n];
}

__global__ void cast_bf16_kernel(__bf16* __restrict__ dst,
                                 const float* __restrict__ src, int n) {
  int i = blockIdx.x * blockDim.x + threadIdx.x;
  if (i < n) dst[i] = (__bf16)src[i];
}

// ---------------------------------------------------------------------------
// Embedding gather: x[b,l,:] = tok_emb[token[b,l],:] + pos_emb[l,:]
// ---------------------------------------------------------------------------
__global__ __launch_bounds__(128)
void embed_kernel(const int* __restrict__ tok,
                  const float* __restrict__ te,
                  const float* __restrict__ pe,
                  float* __restrict__ x) {
  int row = blockIdx.x;                 // b*L + l
  int l   = row & (TC_L - 1);
  int t   = tok[row];
  int d   = threadIdx.x * 4;
  float4 a = *reinterpret_cast<const float4*>(te + (size_t)t * TC_D + d);
  float4 p = *reinterpret_cast<const float4*>(pe + (size_t)l * TC_D + d);
  a.x += p.x; a.y += p.y; a.z += p.z; a.w += p.w;
  *reinterpret_cast<float4*>(x + (size_t)row * TC_D + d) = a;
}

// ---------------------------------------------------------------------------
// LayerNorm (D=512) + cast to bf16.  One wave per row, 8 rows per block.
// remapL!=0: dst row (b*remapL + l) reads src row (b*srcL + l)  (prefix LN).
// ---------------------------------------------------------------------------
__global__ __launch_bounds__(256)
void ln_cast_kernel(const float* __restrict__ x, __bf16* __restrict__ y,
                    const float* __restrict__ w, const float* __restrict__ b,
                    int Nrows, int remapL, int srcL) {
  int row = blockIdx.x * 8 + (threadIdx.x >> 5);
  if (row >= Nrows) return;
  int lane = threadIdx.x & 31;
  int srow = row;
  if (remapL) { int bb = row / remapL; srow = bb * srcL + (row - bb * remapL); }
  const float* xr = x + (size_t)srow * TC_D;

  float4 v[4];
  float s = 0.f, sq = 0.f;
#pragma unroll
  for (int i = 0; i < 4; ++i) {
    v[i] = *reinterpret_cast<const float4*>(xr + lane * 16 + i * 4);
    s  += v[i].x + v[i].y + v[i].z + v[i].w;
    sq += v[i].x * v[i].x + v[i].y * v[i].y + v[i].z * v[i].z + v[i].w * v[i].w;
  }
#pragma unroll
  for (int off = 16; off; off >>= 1) {
    s  += __shfl_xor(s,  off, 32);
    sq += __shfl_xor(sq, off, 32);
  }
  float mean = s * (1.f / TC_D);
  float var  = sq * (1.f / TC_D) - mean * mean;
  float rstd = rsqrtf(var + 1e-5f);

  __bf16* yr = y + (size_t)row * TC_D;
#pragma unroll
  for (int i = 0; i < 4; ++i) {
    int d = lane * 16 + i * 4;
    float4 wv = *reinterpret_cast<const float4*>(w + d);
    float4 bv = *reinterpret_cast<const float4*>(b + d);
    yr[d + 0] = (__bf16)((v[i].x - mean) * rstd * wv.x + bv.x);
    yr[d + 1] = (__bf16)((v[i].y - mean) * rstd * wv.y + bv.y);
    yr[d + 2] = (__bf16)((v[i].z - mean) * rstd * wv.z + bv.z);
    yr[d + 3] = (__bf16)((v[i].w - mean) * rstd * wv.w + bv.w);
  }
}

// ---------------------------------------------------------------------------
// WMMA bf16 GEMM:  out = epilogue(A[M,K] @ Bt[N,K]^T + bias)
//   block 256x64, 8 waves (4 row-groups x 2 col-groups), wave tile 64x32,
//   K step 32, LDS staged (padded rows -> conflict-free ds_load_b128).
// EPI: 0 = exact GeLU -> bf16   1 = residual add f32 (outF += v)
//      2 = tanh -> bf16         3 = plain f32 store
// ---------------------------------------------------------------------------
template <int EPI>
__global__ __launch_bounds__(256)
void gemm_wmma_kernel(const __bf16* __restrict__ A, int lda,
                      const __bf16* __restrict__ Bt, int ldb,
                      int M, int K,
                      const float* __restrict__ bias,
                      float* __restrict__ outF,
                      __bf16* __restrict__ outB, int ldo) {
  __shared__ __align__(16) __bf16 sA[BLK_M * LDS_PAD];
  __shared__ __align__(16) __bf16 sB[BLK_N * LDS_PAD];

  const int tid   = threadIdx.x;
  const int lane  = tid & 31;
  const int wave  = tid >> 5;
  const int half  = lane >> 4;        // 0 | 1
  const int l16   = lane & 15;
  const int rowB  = blockIdx.y * BLK_M;
  const int colB  = blockIdx.x * BLK_N;
  const int waveM = (wave >> 1) * 64; // 0,64,128,192
  const int waveN = (wave & 1) * 32;  // 0,32

  v8f acc[4][2];
#pragma unroll
  for (int m = 0; m < 4; ++m)
#pragma unroll
    for (int n = 0; n < 2; ++n) acc[m][n] = (v8f)(0.f);

  for (int k0 = 0; k0 < K; k0 += 32) {
    // ---- global -> LDS (16B chunks) -----------------------------------
#pragma unroll
    for (int c = 0; c < 4; ++c) {          // A: 256 rows x 4 chunks = 1024
      int chunk = tid + c * 256;
      int r  = chunk >> 2;
      int cc = chunk & 3;
      int grow = imind(rowB + r, M - 1);   // clamp tail rows (M=16376 case)
      const uint4* src = reinterpret_cast<const uint4*>(
          A + (size_t)grow * lda + k0 + cc * 8);
      *reinterpret_cast<uint4*>(&sA[r * LDS_PAD + cc * 8]) = *src;
      if (c == 0 && k0 + 32 < K)           // emits global_prefetch_b8
        __builtin_prefetch(A + (size_t)grow * lda + k0 + 32 + cc * 8, 0, 1);
    }
    {                                      // B: 64 rows x 4 chunks = 256
      int r  = tid >> 2;
      int cc = tid & 3;
      const uint4* src = reinterpret_cast<const uint4*>(
          Bt + (size_t)(colB + r) * ldb + k0 + cc * 8);
      *reinterpret_cast<uint4*>(&sB[r * LDS_PAD + cc * 8]) = *src;
    }
    __syncthreads();

    // ---- fragment loads (ds_load_b128 pairs) --------------------------
    v16bf af[4], bfv[2];
#pragma unroll
    for (int m = 0; m < 4; ++m) {
      const __bf16* p = &sA[(waveM + m * 16 + l16) * LDS_PAD + half * 8];
      v8bf lo = *reinterpret_cast<const v8bf*>(p);
      v8bf hi = *reinterpret_cast<const v8bf*>(p + 16);
#pragma unroll
      for (int i = 0; i < 8; ++i) { af[m][i] = lo[i]; af[m][i + 8] = hi[i]; }
    }
#pragma unroll
    for (int n = 0; n < 2; ++n) {
      const __bf16* p = &sB[(waveN + n * 16 + l16) * LDS_PAD + half * 8];
      v8bf lo = *reinterpret_cast<const v8bf*>(p);
      v8bf hi = *reinterpret_cast<const v8bf*>(p + 16);
#pragma unroll
      for (int i = 0; i < 8; ++i) { bfv[n][i] = lo[i]; bfv[n][i + 8] = hi[i]; }
    }

    // ---- 8x v_wmma_f32_16x16x32_bf16 ----------------------------------
#pragma unroll
    for (int m = 0; m < 4; ++m)
#pragma unroll
      for (int n = 0; n < 2; ++n)
        acc[m][n] = __builtin_amdgcn_wmma_f32_16x16x32_bf16(
            false, af[m], false, bfv[n], (short)0, acc[m][n], false, false);

    __syncthreads();
  }

  // ---- epilogue --------------------------------------------------------
#pragma unroll
  for (int m = 0; m < 4; ++m) {
#pragma unroll
    for (int n = 0; n < 2; ++n) {
#pragma unroll
      for (int r = 0; r < 8; ++r) {
        int row = rowB + waveM + m * 16 + half * 8 + r;
        int col = colB + waveN + n * 16 + l16;
        if (row < M) {
          float v = acc[m][n][r] + (bias ? bias[col] : 0.f);
          size_t o = (size_t)row * ldo + col;
          if (EPI == 0) {                // exact GeLU -> bf16
            float g = 0.5f * v * (1.f + erff(v * 0.70710678118654752f));
            outB[o] = (__bf16)g;
          } else if (EPI == 1) {         // residual accumulate f32
            outF[o] += v;
          } else if (EPI == 2) {         // tanh -> bf16
            outB[o] = (__bf16)tanhf(v);
          } else {                       // plain f32
            outF[o] = v;
          }
        }
      }
    }
  }
}

// ---------------------------------------------------------------------------
// Exact ordered-triplet scan: one thread per (b,c); 2047-step recurrence.
// u layout: [3, NP, 64] f32  (ua, ub, uc)
// ---------------------------------------------------------------------------
__global__ __launch_bounds__(256)
void triplet_scan_kernel(const float* __restrict__ u, float* __restrict__ sbuf) {
  int idx = blockIdx.x * blockDim.x + threadIdx.x;   // 0..511
  if (idx >= TC_B * TC_C) return;
  int b = idx >> 6;
  int c = idx & 63;
  const float* ua = u;
  const float* ub = u + (size_t)TC_NP * TC_C;
  const float* uc = u + (size_t)2 * TC_NP * TC_C;
  size_t base = (size_t)b * (TC_L - 1) * TC_C + c;
  float pa = 0.f, pt = 0.f, s = 0.f;
  for (int l = 0; l < TC_L - 1; ++l) {
    size_t o = base + (size_t)l * TC_C;
    float va = ua[o], vb = ub[o], vc = uc[o];
    s  += vc * pt;       // uses exclusive prefT
    pt += vb * pa;       // uses exclusive prefA
    pa += va;
  }
  sbuf[idx] = s;
}

// ---------------------------------------------------------------------------
// Query head: out[b,c] = s[b,c]/denom + LN(x[b,L-1]) @ Wq[:,c] + bq[c]
// ---------------------------------------------------------------------------
__global__ __launch_bounds__(64)
void query_kernel(const float* __restrict__ x,
                  const float* __restrict__ lnw, const float* __restrict__ lnb,
                  const float* __restrict__ Wq,  const float* __restrict__ bq,
                  const float* __restrict__ sbuf, float* __restrict__ out,
                  float inv_denom) {
  __shared__ float xs[TC_D];
  __shared__ float rs[64], rq[64];
  int b = blockIdx.x;
  int t = threadIdx.x;
  const float* xr = x + ((size_t)b * TC_L + (TC_L - 1)) * TC_D;
  float s = 0.f, sq = 0.f;
  for (int d = t; d < TC_D; d += 64) {
    float v = xr[d];
    xs[d] = v; s += v; sq += v * v;
  }
  rs[t] = s; rq[t] = sq;
  __syncthreads();
  for (int off = 32; off; off >>= 1) {
    if (t < off) { rs[t] += rs[t + off]; rq[t] += rq[t + off]; }
    __syncthreads();
  }
  float mean = rs[0] * (1.f / TC_D);
  float var  = rq[0] * (1.f / TC_D) - mean * mean;
  float rstd = rsqrtf(var + 1e-5f);
  float acc = bq[t];
  for (int d = 0; d < TC_D; ++d) {
    float xn = (xs[d] - mean) * rstd * lnw[d] + lnb[d];
    acc += xn * Wq[(size_t)d * TC_C + t];
  }
  out[(size_t)b * TC_C + t] = sbuf[(size_t)b * TC_C + t] * inv_denom + acc;
}

// ---------------------------------------------------------------------------
// Host launch
// ---------------------------------------------------------------------------
extern "C" void kernel_launch(void* const* d_in, const int* in_sizes, int n_in,
                              void* d_out, int out_size, void* d_ws, size_t ws_size,
                              hipStream_t stream) {
  (void)in_sizes; (void)n_in; (void)out_size; (void)ws_size;
  const int*   tok      = (const int*)  d_in[0];
  const float* tok_emb  = (const float*)d_in[1];
  const float* pos_emb  = (const float*)d_in[2];
  const float* sln_w    = (const float*)d_in[3];
  const float* sln_b    = (const float*)d_in[4];
  const float* sw1      = (const float*)d_in[5];
  const float* sb1      = (const float*)d_in[6];
  const float* sw2      = (const float*)d_in[7];
  const float* sb2      = (const float*)d_in[8];
  const float* rln_w    = (const float*)d_in[9];
  const float* rln_b    = (const float*)d_in[10];
  const float* Wa       = (const float*)d_in[11];
  const float* Wb       = (const float*)d_in[12];
  const float* Wc       = (const float*)d_in[13];
  const float* cls[3]   = {(const float*)d_in[14], (const float*)d_in[15],
                           (const float*)d_in[16]};
  const float* qln_w    = (const float*)d_in[17];
  const float* qln_b    = (const float*)d_in[18];
  const float* Wq       = (const float*)d_in[19];
  const float* bq       = (const float*)d_in[20];
  float* out = (float*)d_out;

  // ---- workspace carve-up (256B aligned) ----
  char* ws = (char*)d_ws;
  size_t off = 0;
  auto carve = [&](size_t bytes) -> char* {
    char* p = ws + off;
    off = (off + bytes + 255) & ~(size_t)255;
    return p;
  };
  float*  x    = (float*) carve((size_t)TC_N * TC_D * 4);   // 32 MB
  __bf16* xb   = (__bf16*)carve((size_t)TC_N * TC_D * 2);   // 16 MB (reused: abc)
  __bf16* hbuf = (__bf16*)carve((size_t)TC_N * TC_H * 2);   // 32 MB (reused: pb + u)
  __bf16* w1t  = (__bf16*)carve((size_t)TC_NBLK * TC_H * TC_D * 2);
  __bf16* w2t  = (__bf16*)carve((size_t)TC_NBLK * TC_D * TC_H * 2);
  __bf16* wabct= (__bf16*)carve((size_t)192 * TC_D * 2);
  __bf16* clsb = (__bf16*)carve((size_t)3 * TC_C * TC_R * 2);
  float*  sbuf = (float*) carve((size_t)TC_B * TC_C * 4);

  __bf16* abc = xb;                                       // [NP,192] bf16
  __bf16* pb  = hbuf;                                     // [NP,512] bf16
  float*  u   = (float*)((char*)hbuf + (size_t)16 * 1024 * 1024); // [3,NP,64] f32

  // ---- weight prep ----
  for (int i = 0; i < TC_NBLK; ++i) {
    int tot1 = TC_H * TC_D;
    transpose_cast_kernel<<<(tot1 + 255) / 256, 256, 0, stream>>>(
        w1t + (size_t)i * tot1, sw1 + (size_t)i * tot1, TC_H, TC_D);
    transpose_cast_kernel<<<(tot1 + 255) / 256, 256, 0, stream>>>(
        w2t + (size_t)i * tot1, sw2 + (size_t)i * tot1, TC_D, TC_H);
  }
  {
    int tot = TC_R * TC_D;  // 64 x 512 per role matrix
    transpose_cast_kernel<<<(tot + 255) / 256, 256, 0, stream>>>(
        wabct + 0 * (size_t)tot, Wa, TC_R, TC_D);
    transpose_cast_kernel<<<(tot + 255) / 256, 256, 0, stream>>>(
        wabct + 1 * (size_t)tot, Wb, TC_R, TC_D);
    transpose_cast_kernel<<<(tot + 255) / 256, 256, 0, stream>>>(
        wabct + 2 * (size_t)tot, Wc, TC_R, TC_D);
  }
  for (int s = 0; s < 3; ++s)
    cast_bf16_kernel<<<(TC_C * TC_R + 255) / 256, 256, 0, stream>>>(
        clsb + (size_t)s * TC_C * TC_R, cls[s], TC_C * TC_R);

  // ---- embedding ----
  embed_kernel<<<TC_N, 128, 0, stream>>>(tok, tok_emb, pos_emb, x);

  // ---- stem blocks ----
  for (int i = 0; i < TC_NBLK; ++i) {
    ln_cast_kernel<<<TC_N / 8, 256, 0, stream>>>(
        x, xb, sln_w + (size_t)i * TC_D, sln_b + (size_t)i * TC_D, TC_N, 0, 0);
    dim3 g1(TC_H / BLK_N, TC_N / BLK_M);
    gemm_wmma_kernel<0><<<g1, 256, 0, stream>>>(                 // GeLU -> h
        xb, TC_D, w1t + (size_t)i * TC_H * TC_D, TC_D, TC_N, TC_D,
        sb1 + (size_t)i * TC_H, nullptr, hbuf, TC_H);
    dim3 g2(TC_D / BLK_N, TC_N / BLK_M);
    gemm_wmma_kernel<1><<<g2, 256, 0, stream>>>(                 // x += h @ w2
        hbuf, TC_H, w2t + (size_t)i * TC_D * TC_H, TC_H, TC_N, TC_H,
        sb2 + (size_t)i * TC_D, x, nullptr, TC_D);
  }

  // ---- role projections ----
  ln_cast_kernel<<<(TC_NP + 7) / 8, 256, 0, stream>>>(
      x, pb, rln_w, rln_b, TC_NP, TC_L - 1, TC_L);
  {
    dim3 g(192 / BLK_N, (TC_NP + BLK_M - 1) / BLK_M);
    gemm_wmma_kernel<2><<<g, 256, 0, stream>>>(                  // tanh -> abc
        pb, TC_D, wabct, TC_D, TC_NP, TC_D, nullptr, nullptr, abc, 192);
  }
  for (int s = 0; s < 3; ++s) {
    dim3 g(1, (TC_NP + BLK_M - 1) / BLK_M);
    gemm_wmma_kernel<3><<<g, 256, 0, stream>>>(                  // u = abc_s @ cls^T
        abc + (size_t)s * 64, 192, clsb + (size_t)s * TC_C * TC_R, TC_R,
        TC_NP, TC_R, nullptr, u + (size_t)s * TC_NP * TC_C, nullptr, TC_C);
  }

  // ---- scan + head ----
  triplet_scan_kernel<<<2, 256, 0, stream>>>(u, sbuf);

  double Lp = (double)(TC_L - 1);
  float inv_denom = (float)(6.0 / (Lp * (Lp - 1.0) * (Lp - 2.0)));
  query_kernel<<<TC_B, 64, 0, stream>>>(x, qln_w, qln_b, Wq, bq, sbuf, out,
                                        inv_denom);
}
